// LSTMActuatorModel_17239998726363
// MI455X (gfx1250) — compile-verified, matrix operand order checked by reading, not run on hardware
//
#include <hip/hip_runtime.h>
#include <hip/hip_bf16.h>

// LSTM (I=2, H=32, gates=128) + FC(32->1), B=2048, T=512.
// One wave32 per 16-batch tile. Per timestep (17 WMMAs):
//   accy    = WMMA_f16(A_h, B_fc, C=b_fc)   // y_{t-1} = h_{t-1}.W_fc + b_fc
//   acc[tt] = WMMA_f32_16x16x4(A_x, B_x, 0) // x @ W_ih^T + (b_ih+b_hh), rank-3, f32
//   acc[tt] = WMMA_f16(A_h, B_hh, acc)      // + h_{t-1} @ W_hh^T (K=32=H)
// All weight tiles live in registers for all T. c stays in registers; h
// round-trips through 1KB LDS with a K-permutation (perm(2u)=u, perm(2u+1)=u+16)
// so each lane stores its two h values as one packed ds_store_b32.
// x loads are software-pipelined one step ahead.

typedef __attribute__((ext_vector_type(16))) _Float16 v16h;
typedef __attribute__((ext_vector_type(8)))  _Float16 v8h;
typedef __attribute__((ext_vector_type(2)))  _Float16 v2h;
typedef __attribute__((ext_vector_type(8)))  float    v8f;
typedef __attribute__((ext_vector_type(2)))  float    v2f;

#define HID   32
#define NTILE 8      // 4*H / 16 gate tiles

#if __has_builtin(__builtin_amdgcn_wmma_f32_16x16x4_f32)
#define HAS_K4 1
#else
#define HAS_K4 0
#endif

#if __has_builtin(__builtin_amdgcn_tanhf)
__device__ __forceinline__ float tanh_(float x) { return __builtin_amdgcn_tanhf(x); }
__device__ __forceinline__ float sigm(float x) {
    return __builtin_fmaf(0.5f, __builtin_amdgcn_tanhf(0.5f * x), 0.5f);
}
#else
__device__ __forceinline__ float fast_rcp(float x) { return __builtin_amdgcn_rcpf(x); }
__device__ __forceinline__ float tanh_(float x) {
    float e = __expf(-2.0f * x);
    return (1.0f - e) * fast_rcp(1.0f + e);
}
__device__ __forceinline__ float sigm(float x) { return fast_rcp(1.0f + __expf(-x)); }
#endif

// K-slot -> hidden-index permutation shared by A_h, B_hh, B_fc.
__device__ __forceinline__ int permK(int k) { return (k & 1) ? ((k >> 1) + 16) : (k >> 1); }

__global__ __launch_bounds__(32)
void lstm_wmma_kernel(const float* __restrict__ x,
                      const float* __restrict__ W_ih,
                      const float* __restrict__ W_hh,
                      const float* __restrict__ b_ih,
                      const float* __restrict__ b_hh,
                      const float* __restrict__ W_fc,
                      const float* __restrict__ b_fc,
                      float* __restrict__ out,
                      int T)
{
    __shared__ _Float16 hbuf[16 * HID];   // h_{t-1}: [batch 16][K-slot 32] (permuted), f16

    const int lane = threadIdx.x;         // 0..31 (wave32)
    const int col  = lane & 15;           // N column / batch row for A_x
    const int hi   = lane >> 4;           // half-wave select
    const int b0   = blockIdx.x * 16;

    // ---- W_hh tiles in WMMA B-operand layout (f16, K permuted), resident all T ----
    v16h Bt[NTILE];
#pragma unroll
    for (int tt = 0; tt < NTILE; ++tt) {
        const float* wrow = W_hh + (size_t)(tt * 16 + col) * HID;
#pragma unroll
        for (int e = 0; e < 8; ++e) {
            Bt[tt][e]     = (_Float16)wrow[permK(hi * 8 + e)];
            Bt[tt][e + 8] = (_Float16)wrow[permK(16 + hi * 8 + e)];
        }
    }

    // ---- input-projection tiles ----
#if HAS_K4
    // f32 16x16x4: lanes 0-15 carry K=0,1 (W_ih cols), lanes 16-31 carry K=2,3 (bias, 0)
    v2f Bx[NTILE];
#pragma unroll
    for (int tt = 0; tt < NTILE; ++tt) {
        int g = tt * 16 + col;
        if (hi == 0) { Bx[tt][0] = W_ih[g * 2 + 0];     Bx[tt][1] = W_ih[g * 2 + 1]; }
        else         { Bx[tt][0] = b_ih[g] + b_hh[g];   Bx[tt][1] = 0.0f; }
    }
#else
    v16h Bx[NTILE];
#pragma unroll
    for (int tt = 0; tt < NTILE; ++tt) {
        Bx[tt] = (v16h){};
        int g = tt * 16 + col;
        if (hi == 0) {
            Bx[tt][0] = (_Float16)W_ih[g * 2 + 0];
            Bx[tt][1] = (_Float16)W_ih[g * 2 + 1];
            Bx[tt][2] = (_Float16)(b_ih[g] + b_hh[g]);
        }
    }
#endif

    // ---- FC tile: only N column 0 populated (lanes 0 and 16), K permuted ----
    v16h Bfc = (v16h){};
    if (col == 0) {
#pragma unroll
        for (int e = 0; e < 8; ++e) {
            Bfc[e]     = (_Float16)W_fc[permK(hi * 8 + e)];
            Bfc[e + 8] = (_Float16)W_fc[permK(16 + hi * 8 + e)];
        }
    }
    // b_fc folded into the y-WMMA C operand (loop-invariant splat)
    v8f Cfc;
    {
        float bfc = b_fc[0];
#pragma unroll
        for (int e = 0; e < 8; ++e) Cfc[e] = bfc;
    }

    // ---- init recurrent state ----
    float c1[8], c2[8];                   // cell state: 8 batch rows x hidden {col, col+16}
#pragma unroll
    for (int r = 0; r < 8; ++r) { c1[r] = 0.f; c2[r] = 0.f; }
    for (int i = lane; i < 16 * HID; i += 32) hbuf[i] = (_Float16)0.f;
    __syncthreads();

    const size_t xrow = (size_t)T * 2;
    const float* pxbase = x + (size_t)(b0 + col) * xrow;

#if !HAS_K4
    v16h Ax = (v16h){};
    if (hi == 0) Ax[2] = (_Float16)1.0f;  // bias-"1" K row, invariant
#endif

    float2 xv = *(const float2*)pxbase;   // x_0, software-pipelined

    for (int t = 0; t < T; ++t) {
        // ---- complete A_x for this step ----
#if HAS_K4
        v2f Ax;
        Ax[0] = (hi == 0) ? xv.x : 1.0f;  // high half = bias row (1, 0)
        Ax[1] = (hi == 0) ? xv.y : 0.0f;
#else
        if (hi == 0) { Ax[0] = (_Float16)xv.x; Ax[1] = (_Float16)xv.y; }
#endif

        // issue next step's x load now; consumed next iteration
        int tn = (t + 1 < T) ? (t + 1) : t;
        float2 xvn = *(const float2*)(pxbase + (size_t)tn * 2);

        // ---- A_h = h_{t-1} in WMMA A layout (16x32 f16, permuted K) from LDS ----
        v16h A;
        {
            v8h alo = *(const v8h*)&hbuf[col * HID + hi * 8];
            v8h ahi = *(const v8h*)&hbuf[col * HID + 16 + hi * 8];
#pragma unroll
            for (int e = 0; e < 8; ++e) { A[e] = alo[e]; A[e + 8] = ahi[e]; }
        }

        // ---- y_{t-1} = h_{t-1} . W_fc + b_fc : one WMMA, result in N column 0 ----
        v8f accy = __builtin_amdgcn_wmma_f32_16x16x32_f16(
            false, A, false, Bfc, (short)0, Cfc, false, false);

        // ---- gates = x W_ih^T + b, then += h_{t-1} W_hh^T ----
        v8f acc[NTILE];
#pragma unroll
        for (int tt = 0; tt < NTILE; ++tt) {
            v8f z = (v8f){};
#if HAS_K4
            acc[tt] = __builtin_amdgcn_wmma_f32_16x16x4_f32(
                false, Ax, false, Bx[tt], (short)0, z, false, false);
#else
            acc[tt] = __builtin_amdgcn_wmma_f32_16x16x32_f16(
                false, Ax, false, Bx[tt], (short)0, z, false, false);
#endif
        }
#pragma unroll
        for (int tt = 0; tt < NTILE; ++tt) {
            acc[tt] = __builtin_amdgcn_wmma_f32_16x16x32_f16(
                false, A, false, Bt[tt], (short)0, acc[tt], false, false);
        }

        // ---- lane-local LSTM cell update (i,f,g,o in tiles 0/2/4/6 and 1/3/5/7) ----
#pragma unroll
        for (int r = 0; r < 8; ++r) {
            float i1 = sigm(acc[0][r]);
            float f1 = sigm(acc[2][r]);
            float g1 = tanh_(acc[4][r]);
            float o1 = sigm(acc[6][r]);
            c1[r] = f1 * c1[r] + i1 * g1;
            float h1 = o1 * tanh_(c1[r]);

            float i2 = sigm(acc[1][r]);
            float f2 = sigm(acc[3][r]);
            float g2 = tanh_(acc[5][r]);
            float o2 = sigm(acc[7][r]);
            c2[r] = f2 * c2[r] + i2 * g2;
            float h2 = o2 * tanh_(c2[r]);

            // permuted layout: (j=col, j=col+16) adjacent -> one packed b32 store
            int row = r + 8 * hi;
            *(v2h*)&hbuf[row * HID + col * 2] = (v2h){(_Float16)h1, (_Float16)h2};
        }
        __syncthreads();  // order h stores vs next iteration's A loads

        // ---- store y_{t-1} from lanes 0/16 ----
        if (t > 0 && col == 0) {
#pragma unroll
            for (int r = 0; r < 8; ++r)
                out[(size_t)(b0 + r + 8 * hi) * T + (t - 1)] = accy[r];
        }
        xv = xvn;
    }

    // ---- epilogue: y_{T-1} from the final h ----
    {
        v16h A;
        v8h alo = *(const v8h*)&hbuf[col * HID + hi * 8];
        v8h ahi = *(const v8h*)&hbuf[col * HID + 16 + hi * 8];
#pragma unroll
        for (int e = 0; e < 8; ++e) { A[e] = alo[e]; A[e + 8] = ahi[e]; }
        v8f accy = __builtin_amdgcn_wmma_f32_16x16x32_f16(
            false, A, false, Bfc, (short)0, Cfc, false, false);
        if (col == 0) {
#pragma unroll
            for (int r = 0; r < 8; ++r)
                out[(size_t)(b0 + r + 8 * hi) * T + (T - 1)] = accy[r];
        }
    }
}

extern "C" void kernel_launch(void* const* d_in, const int* in_sizes, int n_in,
                              void* d_out, int out_size, void* d_ws, size_t ws_size,
                              hipStream_t stream) {
    const float* x    = (const float*)d_in[0];
    const float* W_ih = (const float*)d_in[1];
    const float* W_hh = (const float*)d_in[2];
    const float* b_ih = (const float*)d_in[3];
    const float* b_hh = (const float*)d_in[4];
    const float* W_fc = (const float*)d_in[5];
    const float* b_fc = (const float*)d_in[6];
    float* out = (float*)d_out;

    const int T = 512;
    const int B = out_size / T;          // OUT == 1
    const int nblocks = B / 16;          // 2048/16 = 128 independent waves

    lstm_wmma_kernel<<<dim3(nblocks), dim3(32), 0, stream>>>(
        x, W_ih, W_hh, b_ih, b_hh, W_fc, b_fc, out, T);

    (void)in_sizes; (void)n_in; (void)d_ws; (void)ws_size;
}
